// GPTLanguageModel_2774548873550
// MI455X (gfx1250) — compile-verified, matrix operand order checked by reading, not run on hardware
//
#include <hip/hip_runtime.h>
#include <math.h>

#define Bb   4
#define Tt   1024
#define Vv   50257
#define Ee   1024
#define Hh   16
#define DHd  64
#define MTOT (Bb * Tt)          // 4096
#define VPAD 50432              // 197 * 256, zero-padded LM weight rows

typedef __attribute__((ext_vector_type(16))) _Float16 v16h;
typedef __attribute__((ext_vector_type(8)))  _Float16 v8h;
typedef __attribute__((ext_vector_type(8)))  float    v8f;

// ---------------------------------------------------------------------------
// WMMA helpers (CDNA5 wave32, 16x16x32 f16 -> f32)
// A fragment: lane<16 -> row M=lane, K {0..7,16..23}; lane>=16 -> row M=lane-16,
// K {8..15,24..31}. B fragment mirrors with lane = N column. So loading from a
// row-major [rows, ld] matrix (rows = M for A, rows = N for B-transposed) is
// two contiguous 16-byte loads per lane.
// ---------------------------------------------------------------------------
__device__ __forceinline__ v16h load_frag(const _Float16* p, int ld) {
  int lane = threadIdx.x & 31;
  int r    = lane & 15;
  int ko   = (lane >> 4) << 3;            // 0 or 8
  const _Float16* q = p + (long)r * ld + ko;
  v8h lo = *(const v8h*)q;                // K = ko .. ko+7
  v8h hi = *(const v8h*)(q + 16);         // K = ko+16 .. ko+23
  v16h out;
#pragma unroll
  for (int i = 0; i < 8; ++i) { out[i] = lo[i]; out[i + 8] = hi[i]; }
  return out;
}

__device__ __forceinline__ v8f wmma_f16(v16h a, v16h b, v8f c) {
  return __builtin_amdgcn_wmma_f32_16x16x32_f16(false, a, false, b,
                                                (short)0, c, false, false);
}

__device__ __forceinline__ v8f zero8() {
  v8f z;
#pragma unroll
  for (int i = 0; i < 8; ++i) z[i] = 0.0f;
  return z;
}

// ---------------------------------------------------------------------------
// 1) Embedding: x_f16[b*T+t][e] = f16(tok_emb[idx[b,t]][e] + pos_emb[t][e])
// ---------------------------------------------------------------------------
__global__ void embed_kernel(const int* __restrict__ idx,
                             const float* __restrict__ tok,
                             const float* __restrict__ pos,
                             _Float16* __restrict__ x) {
  int bt = blockIdx.x;                    // 0..4095
  int t  = bt & (Tt - 1);
  long  tokoff = (long)idx[bt] * Ee;
  const float* te = tok + tokoff;
  const float* pe = pos + (long)t * Ee;
  for (int e = threadIdx.x; e < Ee; e += blockDim.x)
    x[(long)bt * Ee + e] = (_Float16)(te[e] + pe[e]);
}

// ---------------------------------------------------------------------------
// 2a) Wq/Wk/Wv [H,E,DH] f32 -> Wt [H*DH, E] f16 (B-transposed layout)
// ---------------------------------------------------------------------------
__global__ void conv_qkv_w(const float* __restrict__ W, _Float16* __restrict__ Wt) {
  int n = blockIdx.x;                     // 0..1023 = h*64 + d
  int h = n >> 6, d = n & 63;
  const float* src = W + (long)h * Ee * DHd + d;   // W[h][e][d], stride DHd over e
  for (int e = threadIdx.x; e < Ee; e += blockDim.x)
    Wt[(long)n * Ee + e] = (_Float16)src[(long)e * DHd];
}

// ---------------------------------------------------------------------------
// 2b) W_lm [E, V] f32 -> Wlm_t [VPAD, E] f16, LDS-tiled transpose, zero pad
// ---------------------------------------------------------------------------
__global__ void transpose_lm(const float* __restrict__ W, _Float16* __restrict__ Wt) {
  __shared__ float tile[32][33];
  int n0 = blockIdx.x * 32;               // output row / input col
  int e0 = blockIdx.y * 32;               // input row / output col
  int tx = threadIdx.x, ty = threadIdx.y; // 32 x 8
  for (int i = ty; i < 32; i += 8) {
    int n = n0 + tx;
    tile[i][tx] = (n < Vv) ? W[(long)(e0 + i) * Vv + n] : 0.0f;
  }
  __syncthreads();
  for (int i = ty; i < 32; i += 8)
    Wt[(long)(n0 + i) * Ee + (e0 + tx)] = (_Float16)tile[tx][i];
}

// ---------------------------------------------------------------------------
// 3) QKV GEMM: A [4096,1024] f16 x Bt [1024,1024] f16 -> f16
//    MODE 0: out row-major [M,N] * scale   (q with 1/sqrt(64), k with 1.0)
//    MODE 1: out = v_t layout [B,H,DH,T]   (contiguous 16B stores per lane)
//    Block = 8 waves; wave computes 32M x 64N; block tile 64M x 256N.
//    Fragments loaded at point of use: keeps VGPR pressure low (no spills);
//    latency hidden by resident waves.
// ---------------------------------------------------------------------------
template <int MODE>
__global__ void gemm_qkv_kernel(const _Float16* __restrict__ A,
                                const _Float16* __restrict__ Bt,
                                _Float16* __restrict__ O, float scale) {
  const int N = Hh * DHd;                 // 1024
  int wid  = threadIdx.x >> 5;
  int lane = threadIdx.x & 31;
  int hi8  = (lane & 16) >> 1;            // 0 or 8
  int nc   = lane & 15;
  int m0 = blockIdx.y * 64  + (wid >> 2) * 32;
  int n0 = blockIdx.x * 256 + (wid & 3) * 64;

  v8f acc[2][4];
#pragma unroll
  for (int i = 0; i < 2; ++i)
#pragma unroll
    for (int j = 0; j < 4; ++j) acc[i][j] = zero8();

  for (int kk = 0; kk < Ee; kk += 32) {
    v16h a0 = load_frag(A + (long)m0 * Ee + kk, Ee);
    v16h a1 = load_frag(A + (long)(m0 + 16) * Ee + kk, Ee);
#pragma unroll
    for (int j = 0; j < 4; ++j) {
      v16h bf = load_frag(Bt + (long)(n0 + j * 16) * Ee + kk, Ee);
      acc[0][j] = wmma_f16(a0, bf, acc[0][j]);
      acc[1][j] = wmma_f16(a1, bf, acc[1][j]);
    }
  }

  if (MODE == 0) {
#pragma unroll
    for (int i2 = 0; i2 < 2; ++i2)
#pragma unroll
      for (int j = 0; j < 4; ++j) {
        int n = n0 + j * 16 + nc;
#pragma unroll
        for (int e = 0; e < 8; ++e)
          O[(long)(m0 + i2 * 16 + e + hi8) * N + n] = (_Float16)(acc[i2][j][e] * scale);
      }
  } else {
    int b = m0 >> 10;
#pragma unroll
    for (int i2 = 0; i2 < 2; ++i2)
#pragma unroll
      for (int j = 0; j < 4; ++j) {
        int n  = n0 + j * 16 + nc;          // = h*64 + d
        int t0 = (m0 & (Tt - 1)) + i2 * 16 + hi8;  // 8 consecutive t's
        v8h pk;
#pragma unroll
        for (int e = 0; e < 8; ++e) pk[e] = (_Float16)acc[i2][j][e];
        *(v8h*)(O + (long)(b * 1024 + n) * Tt + t0) = pk;   // vt row = (b*H+h)*64+d
      }
  }
}

// ---------------------------------------------------------------------------
// 4) Flash attention (causal, online softmax). grid (T/64, B*H), block 128.
//    Wave owns 16 q-rows; streams keys in steps of 32 (two 16x16 S tiles ->
//    one K=32 P fragment via LDS transpose); O accum 16x64 in 4 v8f.
//    Shuffle reductions are step-major: 8 independent ds_bpermute chains
//    issue per butterfly step (one wait each), instead of 32 serialized ones.
//    K/V fragments loaded at use to stay spill-free.
// ---------------------------------------------------------------------------
__global__ void attn_kernel(const _Float16* __restrict__ Q,
                            const _Float16* __restrict__ Km,
                            const _Float16* __restrict__ Vt,
                            _Float16* __restrict__ O16) {
  __shared__ _Float16 plds[4 * 16 * 40];  // per-wave 16x40 (padded) P staging
  int wid  = threadIdx.x >> 5;
  int lane = threadIdx.x & 31;
  int hi8  = (lane & 16) >> 1;
  int nc   = lane & 15;
  int bh = blockIdx.y, b = bh >> 4, h = bh & 15;
  int qt0 = blockIdx.x * 64 + wid * 16;
  int qi  = qt0 >> 4;

  const _Float16* qbase = Q + (long)(b * Tt + qt0) * Ee + h * DHd;
  v16h qA0 = load_frag(qbase, Ee);        // d = 0..31
  v16h qA1 = load_frag(qbase + 32, Ee);   // d = 32..63
  const _Float16* vtb = Vt + (long)bh * DHd * Tt;
  _Float16* lw = plds + wid * 640;

  float mrow[8], lrow[8];
  v8f acc[4];
#pragma unroll
  for (int e = 0; e < 8; ++e) { mrow[e] = -1e30f; lrow[e] = 0.0f; }
#pragma unroll
  for (int ds = 0; ds < 4; ++ds) acc[ds] = zero8();

  int nsteps = (qi + 2) >> 1;             // ceil((qi+1)/2)
  for (int j = 0; j < nsteps; ++j) {
    int st0 = j * 32, st1 = st0 + 16;
    const _Float16* kp0 = Km + (long)(b * Tt + st0) * Ee + h * DHd;
    const _Float16* kp1 = Km + (long)(b * Tt + st1) * Ee + h * DHd;
    v8f s0 = zero8(), s1 = zero8();
    s0 = wmma_f16(qA0, load_frag(kp0, Ee), s0);
    s0 = wmma_f16(qA1, load_frag(kp0 + 32, Ee), s0);
    s1 = wmma_f16(qA0, load_frag(kp1, Ee), s1);
    s1 = wmma_f16(qA1, load_frag(kp1 + 32, Ee), s1);

    if (st0 + 15 > qt0) {                 // causal mask (q already scaled by 1/8)
      int sc = st0 + nc;
#pragma unroll
      for (int e = 0; e < 8; ++e)
        if (sc > qt0 + e + hi8) s0[e] = -1e30f;
    }
    if (st1 + 15 > qt0) {
      int sc = st1 + nc;
#pragma unroll
      for (int e = 0; e < 8; ++e)
        if (sc > qt0 + e + hi8) s1[e] = -1e30f;
    }

    // row max over this step's 32 keys: step-major butterfly (8 rows batched)
    float tm[8];
#pragma unroll
    for (int e = 0; e < 8; ++e) tm[e] = fmaxf(s0[e], s1[e]);
#pragma unroll
    for (int off = 1; off <= 8; off <<= 1)
#pragma unroll
      for (int e = 0; e < 8; ++e)
        tm[e] = fmaxf(tm[e], __shfl_xor(tm[e], off, 32));

    float al[8];
#pragma unroll
    for (int e = 0; e < 8; ++e) {
      float nm = fmaxf(mrow[e], tm[e]);
      al[e] = __expf(mrow[e] - nm);
      mrow[e] = nm;
    }

    float rs[8];
#pragma unroll
    for (int e = 0; e < 8; ++e) {
      float p0 = __expf(s0[e] - mrow[e]);
      float p1 = __expf(s1[e] - mrow[e]);
      lw[(e + hi8) * 40 + nc]      = (_Float16)p0;   // C-layout -> A-layout via LDS
      lw[(e + hi8) * 40 + nc + 16] = (_Float16)p1;
      rs[e] = p0 + p1;
    }
#pragma unroll
    for (int off = 1; off <= 8; off <<= 1)
#pragma unroll
      for (int e = 0; e < 8; ++e)
        rs[e] += __shfl_xor(rs[e], off, 32);
#pragma unroll
    for (int e = 0; e < 8; ++e) {
      lrow[e] = lrow[e] * al[e] + rs[e];
#pragma unroll
      for (int ds = 0; ds < 4; ++ds) acc[ds][e] *= al[e];
    }

    asm volatile("s_wait_dscnt 0" ::: "memory");
    v16h pF = load_frag(lw, 40);          // P [16 rows, 32 keys]
#pragma unroll
    for (int ds = 0; ds < 4; ++ds) {
      v16h vB = load_frag(vtb + (long)(ds * 16) * Tt + st0, Tt);
      acc[ds] = wmma_f16(pF, vB, acc[ds]);
    }
  }

#pragma unroll
  for (int e = 0; e < 8; ++e) {
    float inv = 1.0f / lrow[e];
    long rowoff = (long)(b * Tt + qt0 + e + hi8) * Ee + h * DHd;
#pragma unroll
    for (int ds = 0; ds < 4; ++ds)
      O16[rowoff + ds * 16 + nc] = (_Float16)(acc[ds][e] * inv);
  }
}

// ---------------------------------------------------------------------------
// 5) LM head: attn [4096,1024] f16 x Wlm_t [VPAD,1024] f16 -> logits f32 + bias
//    grid (197, 64), block 256. Wlm_t (~103 MB f16) is L2-resident (192 MB).
// ---------------------------------------------------------------------------
__global__ void lm_head_kernel(const _Float16* __restrict__ A,
                               const _Float16* __restrict__ Bt,
                               const float* __restrict__ bias,
                               float* __restrict__ out) {
  int wid  = threadIdx.x >> 5;
  int lane = threadIdx.x & 31;
  int hi8  = (lane & 16) >> 1;
  int nc   = lane & 15;
  int m0 = blockIdx.y * 64  + (wid >> 2) * 32;
  int n0 = blockIdx.x * 256 + (wid & 3) * 64;

  v8f acc[2][4];
#pragma unroll
  for (int i = 0; i < 2; ++i)
#pragma unroll
    for (int j = 0; j < 4; ++j) acc[i][j] = zero8();

  for (int kk = 0; kk < Ee; kk += 32) {
    v16h a0 = load_frag(A + (long)m0 * Ee + kk, Ee);
    v16h a1 = load_frag(A + (long)(m0 + 16) * Ee + kk, Ee);
#pragma unroll
    for (int j = 0; j < 4; ++j) {
      v16h bf = load_frag(Bt + (long)(n0 + j * 16) * Ee + kk, Ee);
      acc[0][j] = wmma_f16(a0, bf, acc[0][j]);
      acc[1][j] = wmma_f16(a1, bf, acc[1][j]);
    }
  }
#pragma unroll
  for (int i2 = 0; i2 < 2; ++i2)
#pragma unroll
    for (int j = 0; j < 4; ++j) {
      int n = n0 + j * 16 + nc;
      if (n < Vv) {
        float bv = bias[n];
#pragma unroll
        for (int e = 0; e < 8; ++e)
          out[(long)(m0 + i2 * 16 + e + hi8) * Vv + n] = acc[i2][j][e] + bv;
      }
    }
}

// ---------------------------------------------------------------------------
// 6) Cross entropy: per-row (4096 blocks) max + sumexp reductions over V.
// ---------------------------------------------------------------------------
__global__ void loss_init(float* __restrict__ loss) { *loss = 0.0f; }

__global__ void loss_kernel(const float* __restrict__ logits,
                            const int* __restrict__ tgt,
                            float* __restrict__ loss) {
  __shared__ float red[256];
  long m = blockIdx.x;
  const float* row = logits + m * (long)Vv;
  int tid = threadIdx.x;

  float mx = -3.4e38f;
  for (int i = tid; i < Vv; i += 256) mx = fmaxf(mx, row[i]);
  red[tid] = mx; __syncthreads();
  for (int s = 128; s > 0; s >>= 1) {
    if (tid < s) red[tid] = fmaxf(red[tid], red[tid + s]);
    __syncthreads();
  }
  mx = red[0]; __syncthreads();

  float sum = 0.0f;
  for (int i = tid; i < Vv; i += 256) sum += __expf(row[i] - mx);
  red[tid] = sum; __syncthreads();
  for (int s = 128; s > 0; s >>= 1) {
    if (tid < s) red[tid] += red[tid + s];
    __syncthreads();
  }
  if (tid == 0) {
    float nll = __logf(red[0]) + mx - row[tgt[m]];
    atomicAdd(loss, nll * (1.0f / (float)MTOT));
  }
}

// ---------------------------------------------------------------------------
extern "C" void kernel_launch(void* const* d_in, const int* in_sizes, int n_in,
                              void* d_out, int out_size, void* d_ws, size_t ws_size,
                              hipStream_t stream) {
  (void)in_sizes; (void)n_in; (void)out_size; (void)ws_size;
  const int*   idx = (const int*)d_in[0];
  const int*   tgt = (const int*)d_in[1];
  const float* tok = (const float*)d_in[2];
  const float* pos = (const float*)d_in[3];
  const float* Wq  = (const float*)d_in[4];
  const float* Wk  = (const float*)d_in[5];
  const float* Wv  = (const float*)d_in[6];
  const float* Wlm = (const float*)d_in[7];
  const float* blm = (const float*)d_in[8];
  float* out = (float*)d_out;

  _Float16* x16   = (_Float16*)d_ws;                 // [4096,1024]
  _Float16* q16   = x16  + (long)MTOT * Ee;          // [4096,1024] (pre-scaled)
  _Float16* k16   = q16  + (long)MTOT * Ee;          // [4096,1024]
  _Float16* vt16  = k16  + (long)MTOT * Ee;          // [B*H*64, 1024] transposed
  _Float16* a16   = vt16 + (long)MTOT * Ee;          // attention out [4096,1024]
  _Float16* wq_t  = a16  + (long)MTOT * Ee;          // [1024,1024]
  _Float16* wk_t  = wq_t + (long)Ee * Ee;
  _Float16* wv_t  = wk_t + (long)Ee * Ee;
  _Float16* wlm_t = wv_t + (long)Ee * Ee;            // [VPAD,1024]

  embed_kernel<<<MTOT, 256, 0, stream>>>(idx, tok, pos, x16);
  conv_qkv_w<<<Ee, 256, 0, stream>>>(Wq, wq_t);
  conv_qkv_w<<<Ee, 256, 0, stream>>>(Wk, wk_t);
  conv_qkv_w<<<Ee, 256, 0, stream>>>(Wv, wv_t);
  transpose_lm<<<dim3(VPAD / 32, Ee / 32), dim3(32, 8), 0, stream>>>(Wlm, wlm_t);

  gemm_qkv_kernel<0><<<dim3(4, 64), 256, 0, stream>>>(x16, wq_t, q16, 0.125f); // 1/sqrt(64)
  gemm_qkv_kernel<0><<<dim3(4, 64), 256, 0, stream>>>(x16, wk_t, k16, 1.0f);
  gemm_qkv_kernel<1><<<dim3(4, 64), 256, 0, stream>>>(x16, wv_t, vt16, 1.0f);

  attn_kernel<<<dim3(Tt / 64, Bb * Hh), 128, 0, stream>>>(q16, k16, vt16, a16);

  lm_head_kernel<<<dim3(VPAD / 256, MTOT / 64), 256, 0, stream>>>(a16, wlm_t, blm, out);

  float* loss = out + (long)MTOT * Vv;
  loss_init<<<1, 1, 0, stream>>>(loss);
  loss_kernel<<<MTOT, 256, 0, stream>>>(out, tgt, loss);
}
